// Model_73409581023752
// MI455X (gfx1250) — matrix operand order repaired and checked by reading, not compiled, not run on hardware
//
#include <hip/hip_runtime.h>

// ---------------------------------------------------------------------------
// MI455X / gfx1250 implementation.
// Phase 1: point MLP (12->16->16->4, ELU) via v_wmma_f32_16x16x32_f16.
//          Chain computed transposed (weights in A-frag, activations in
//          B-frag).  Each wave processes 32 points/iteration with TWO B
//          fragments; all cross-half data movement uses v_permlanex16_b32
//          (VALU, no LDS round-trip / dscnt stalls).
//          Scatter-max uses native global_atomic_max_num_f32 @ device scope.
// Phase 2: 7x (conv3d + BatchNorm(batch stats) + ELU) on tiny 8^3 grids,
//          64 blocks per conv + an 8-block BN reduction kernel per layer.
// ---------------------------------------------------------------------------

typedef __attribute__((ext_vector_type(16))) _Float16 v16h;
typedef __attribute__((ext_vector_type(8)))  float    v8f;

#define GCELLS 15
#define GZF    15.0f
#define NCELL  (GCELLS * GCELLS * GCELLS)   // 3375
#define NB     4
#define NPTS   300000
#define SCENE_CH 12
#define SCENE_SZ (NB * SCENE_CH * NCELL)    // 162000 floats
#define ACT_SZ   (NB * 8 * 512)             // 16384 floats

union Frag16 {
    v16h v;
    _Float16 h[16];
    unsigned u[8];
};

// Cross-half swap (lane i <-> lane i^16) in the VALU: v_permlanex16_b32 with
// identity nibble selects.  Avoids the ds_bpermute + s_wait_dscnt path that
// __shfl_xor(x,16) takes.
__device__ __forceinline__ unsigned xhalf_u32(unsigned x) {
    return (unsigned)__builtin_amdgcn_permlanex16(
        (int)x, (int)x, 0x76543210, 0xfedcba98, false, false);
}
__device__ __forceinline__ float xhalf_f32(float x) {
    union { float f; int i; } a, r;
    a.f = x;
    r.i = __builtin_amdgcn_permlanex16(a.i, a.i, 0x76543210, 0xfedcba98,
                                       false, false);
    return r.f;
}

// ELU.  exp only evaluated on x<=0 where exp2(x*log2e) cannot overflow, so the
// raw v_exp_f32 (no NaN/overflow fixup code) is sufficient.
__device__ __forceinline__ float eluf(float x) {
    return x > 0.0f ? x : (__builtin_amdgcn_exp2f(x * 1.44269504088896f) - 1.0f);
}

// Raw v_sqrt_f32: feature norms are in [0, ~28], no denormal/range fixups
// (the precise sqrtf lowering adds v_cmp_class + cndmask chains).
__device__ __forceinline__ float fast_sqrtf(float x) {
    return __builtin_amdgcn_sqrtf(x);
}

// Native CDNA5 float atomic max (no return), device scope: the scene grid is
// shared across WGPs so the RMW must be performed at the device-coherent L2.
__device__ __forceinline__ void atomic_max_f32(float* p, float v) {
    asm volatile("global_atomic_max_num_f32 %0, %1, off scope:SCOPE_DEV"
                 :: "v"(p), "v"(v)
                 : "memory");
}

// --------------------------------- zero ------------------------------------
__global__ void zero_kernel(float* __restrict__ p, int n) {
    int i = blockIdx.x * blockDim.x + threadIdx.x;
    if (i < n) p[i] = 0.0f;
}

// ------------------------------- encoder -----------------------------------
// gridDim.y = 3 selects the cloud: 0 -> inputs (ch 0..3), 1 -> goals (ch 4..7),
// 2 -> backgrounds (ch 8..11).  Each wave handles 32 points per iteration.
__global__ void __launch_bounds__(256)
encode_kernel(const float* __restrict__ p_inputs,
              const float* __restrict__ p_goals,
              const float* __restrict__ p_backgrounds,
              const float* __restrict__ W1, const float* __restrict__ b1,
              const float* __restrict__ W2, const float* __restrict__ b2,
              const float* __restrict__ W3, const float* __restrict__ b3,
              float* __restrict__ grid)
{
    const float* pts =
        (blockIdx.y == 0) ? p_inputs : (blockIdx.y == 1) ? p_goals : p_backgrounds;
    const int choff = blockIdx.y * 4;

    const int lane = threadIdx.x & 31;
    const int wave = threadIdx.x >> 5;
    const int M    = lane & 15;      // fragment row (hidden-unit index)
    const int hi   = lane >> 4;      // lane-half selector

    // ---- A fragments: transposed weights, padded to 16x32 f16 ----
    // A layout (contiguous K): lane<16 holds K=0..15 in elements 0..15;
    // lane>=16 holds K=16..31.  All K>=K_real padded with zeros.
    Frag16 A1, A2, A3;
    #pragma unroll
    for (int j = 0; j < 4; ++j) {
        const int k0 = 2 * j;
        const int k1 = 8 + 2 * j;
        const bool lo = (hi == 0);
        // layer 1: A1[m][k] = W1[k][m], k < 12          (W1 is [12][16])
        A1.h[2*j]   = (lo && k0     < 12) ? (_Float16)W1[k0 * 16 + M]       : (_Float16)0.0f;
        A1.h[2*j+1] = (lo && k0 + 1 < 12) ? (_Float16)W1[(k0 + 1) * 16 + M] : (_Float16)0.0f;
        A1.h[8+2*j] = (lo && k1     < 12) ? (_Float16)W1[k1 * 16 + M]       : (_Float16)0.0f;
        A1.h[9+2*j] = (lo && k1 + 1 < 12) ? (_Float16)W1[(k1 + 1) * 16 + M] : (_Float16)0.0f;
        // layer 2: A2[m][k] = W2[k][m], k < 16          (W2 is [16][16])
        A2.h[2*j]   = lo ? (_Float16)W2[k0 * 16 + M]       : (_Float16)0.0f;
        A2.h[2*j+1] = lo ? (_Float16)W2[(k0 + 1) * 16 + M] : (_Float16)0.0f;
        A2.h[8+2*j] = lo ? (_Float16)W2[k1 * 16 + M]       : (_Float16)0.0f;
        A2.h[9+2*j] = lo ? (_Float16)W2[(k1 + 1) * 16 + M] : (_Float16)0.0f;
        // layer 3: A3[m][k] = W3[k][m] for m<4 else 0   (W3 is [16][4])
        A3.h[2*j]   = (lo && M < 4) ? (_Float16)W3[k0 * 4 + M]       : (_Float16)0.0f;
        A3.h[2*j+1] = (lo && M < 4) ? (_Float16)W3[(k0 + 1) * 4 + M] : (_Float16)0.0f;
        A3.h[8+2*j] = (lo && M < 4) ? (_Float16)W3[k1 * 4 + M]       : (_Float16)0.0f;
        A3.h[9+2*j] = (lo && M < 4) ? (_Float16)W3[(k1 + 1) * 4 + M] : (_Float16)0.0f;
    }

    // per-lane bias slices matching the D layout (reg r -> row r + 8*hi)
    float bb1[8], bb2[8], bb3[4];
    #pragma unroll
    for (int r = 0; r < 8; ++r) {
        bb1[r] = b1[r + 8 * hi];
        bb2[r] = b2[r + 8 * hi];
    }
    #pragma unroll
    for (int r = 0; r < 4; ++r) bb3[r] = b3[r];

    const int tiles   = (NB * NPTS) / 32;         // 37500 (32 points per tile)
    const int wstride = gridDim.x * (256 / 32);

    for (int tile = blockIdx.x * (256 / 32) + wave; tile < tiles; tile += wstride) {
        // ---- every lane owns one point ----
        const int t = tile * 32 + lane;
        const float px = pts[3 * t + 0];
        const float py = pts[3 * t + 1];
        const float pz = pts[3 * t + 2];
        const float cx = floorf(fminf(fmaxf(px, 0.0f), GZF - 1e-4f));
        const float cy = floorf(fminf(fmaxf(py, 0.0f), GZF - 1e-4f));
        const float cz = floorf(fminf(fmaxf(pz, 0.0f), GZF - 1e-4f));

        Frag16 F;
        {
            float f[12];
            f[0] = px - (cx + 0.5f); f[1] = py - (cy + 0.5f); f[2] = pz - (cz + 0.5f);
            f[3] = px - cx;          f[4] = py - cy;          f[5] = pz - cz;
            f[6] = px - (cx + 1.f);  f[7] = py - (cy + 1.f);  f[8] = pz - (cz + 1.f);
            f[9]  = fast_sqrtf(f[0]*f[0] + f[1]*f[1] + f[2]*f[2]);
            f[10] = fast_sqrtf(f[3]*f[3] + f[4]*f[4] + f[5]*f[5]);
            f[11] = fast_sqrtf(f[6]*f[6] + f[7]*f[7] + f[8]*f[8]);
            #pragma unroll
            for (int i = 0; i < 12; ++i) F.h[i] = (_Float16)f[i];
            F.u[6] = 0u; F.u[7] = 0u;
        }

        // B fragments: frag a = points tile*32+0..15, frag b = +16..31.
        // Lane n<16 needs features of its point (own) / of lane n+16 (xor-16).
        Frag16 Ba, Bb;
        #pragma unroll
        for (int j = 0; j < 8; ++j) { Ba.u[j] = 0u; Bb.u[j] = 0u; }
        #pragma unroll
        for (int j = 0; j < 6; ++j) {
            unsigned o = xhalf_u32(F.u[j]);
            if (hi == 0) { Ba.u[j] = F.u[j]; Bb.u[j] = o; }
        }

        // ---- layer 1: H1^T = W1^T * X^T ----
        v8f c0 = {};
        v8f d1a = __builtin_amdgcn_wmma_f32_16x16x32_f16(
            false, A1.v, false, Ba.v, (short)0, c0, false, false);
        v8f d1b = __builtin_amdgcn_wmma_f32_16x16x32_f16(
            false, A1.v, false, Bb.v, (short)0, c0, false, false);

        Frag16 P1a, P1b;
        #pragma unroll
        for (int r = 0; r < 8; ++r) {
            P1a.h[r] = (_Float16)eluf(d1a[r] + bb1[r]);
            P1b.h[r] = (_Float16)eluf(d1b[r] + bb1[r]);
        }

        // assemble layer-2 B fragments (hidden 8..15 live on lane+16)
        Frag16 B2a, B2b;
        #pragma unroll
        for (int j = 0; j < 8; ++j) { B2a.u[j] = 0u; B2b.u[j] = 0u; }
        #pragma unroll
        for (int j = 0; j < 4; ++j) {
            unsigned oa = xhalf_u32(P1a.u[j]);
            unsigned ob = xhalf_u32(P1b.u[j]);
            if (hi == 0) {
                B2a.u[j] = P1a.u[j]; B2a.u[4 + j] = oa;
                B2b.u[j] = P1b.u[j]; B2b.u[4 + j] = ob;
            }
        }

        // ---- layer 2 ----
        v8f d2a = __builtin_amdgcn_wmma_f32_16x16x32_f16(
            false, A2.v, false, B2a.v, (short)0, c0, false, false);
        v8f d2b = __builtin_amdgcn_wmma_f32_16x16x32_f16(
            false, A2.v, false, B2b.v, (short)0, c0, false, false);

        Frag16 P2a, P2b;
        #pragma unroll
        for (int r = 0; r < 8; ++r) {
            P2a.h[r] = (_Float16)eluf(d2a[r] + bb2[r]);
            P2b.h[r] = (_Float16)eluf(d2b[r] + bb2[r]);
        }

        Frag16 B3a, B3b;
        #pragma unroll
        for (int j = 0; j < 8; ++j) { B3a.u[j] = 0u; B3b.u[j] = 0u; }
        #pragma unroll
        for (int j = 0; j < 4; ++j) {
            unsigned oa = xhalf_u32(P2a.u[j]);
            unsigned ob = xhalf_u32(P2b.u[j]);
            if (hi == 0) {
                B3a.u[j] = P2a.u[j]; B3a.u[4 + j] = oa;
                B3b.u[j] = P2b.u[j]; B3b.u[4 + j] = ob;
            }
        }

        // ---- layer 3 (4 valid output rows) ----
        v8f d3a = __builtin_amdgcn_wmma_f32_16x16x32_f16(
            false, A3.v, false, B3a.v, (short)0, c0, false, false);
        v8f d3b = __builtin_amdgcn_wmma_f32_16x16x32_f16(
            false, A3.v, false, B3b.v, (short)0, c0, false, false);

        // outputs for frag b live on lanes 0..15 -> move to lanes 16..31 so
        // every lane scatters its own point (4 atomics each, full wave).
        float vout[4];
        #pragma unroll
        for (int m = 0; m < 4; ++m) {
            const float va  = d3a[m] + bb3[m];
            const float vb  = d3b[m] + bb3[m];
            const float vbx = xhalf_f32(vb);
            vout[m] = hi ? vbx : va;
        }

        const int b    = t / NPTS;
        const int flat = ((int)cx * GCELLS + (int)cy) * GCELLS + (int)cz;
        float* gbase = grid + (b * SCENE_CH + choff) * NCELL + flat;
        #pragma unroll
        for (int m = 0; m < 4; ++m) {
            atomic_max_f32(gbase + m * NCELL, vout[m]);
        }
    }
}

// ------------------------------ conv stack ---------------------------------
// conv1: [B,12,15,15,15] -> [B,8,8,8,8], k=5, stride=2, pad=2.  1 output/thread.
__global__ void __launch_bounds__(256)
conv1_kernel(const float* __restrict__ in, const float* __restrict__ w,
             const float* __restrict__ bias, float* __restrict__ out)
{
    const int o  = blockIdx.x * blockDim.x + threadIdx.x;   // [B][8][512]
    const int b  = o >> 12;
    const int c  = (o >> 9) & 7;
    const int s  = o & 511;
    const int od = s >> 6, oh = (s >> 3) & 7, ow = s & 7;

    float acc = bias[c];
    for (int ci = 0; ci < 12; ++ci) {
        const float* ib = in + (b * 12 + ci) * NCELL;
        const float* wc = w + (c * 12 + ci) * 125;
        for (int kd = 0; kd < 5; ++kd) {
            const int id = od * 2 - 2 + kd;
            if (id < 0 || id >= 15) continue;
            for (int kh = 0; kh < 5; ++kh) {
                const int ih = oh * 2 - 2 + kh;
                if (ih < 0 || ih >= 15) continue;
                #pragma unroll
                for (int kw = 0; kw < 5; ++kw) {
                    const int iw = ow * 2 - 2 + kw;
                    if (iw < 0 || iw >= 15) continue;
                    acc += ib[id * 225 + ih * 15 + iw] * wc[(kd * 5 + kh) * 5 + kw];
                }
            }
        }
    }
    out[o] = acc;
}

// convN: [B,8,8,8,8] -> [B,8,8,8,8], k=5, stride=1, pad=2.
__global__ void __launch_bounds__(256)
convN_kernel(const float* __restrict__ in, const float* __restrict__ w,
             const float* __restrict__ bias, float* __restrict__ out)
{
    const int o  = blockIdx.x * blockDim.x + threadIdx.x;
    const int b  = o >> 12;
    const int c  = (o >> 9) & 7;
    const int s  = o & 511;
    const int od = s >> 6, oh = (s >> 3) & 7, ow = s & 7;

    float acc = bias[c];
    for (int ci = 0; ci < 8; ++ci) {
        const float* ib = in + (b * 8 + ci) * 512;
        const float* wc = w + (c * 8 + ci) * 125;
        for (int kd = 0; kd < 5; ++kd) {
            const int id = od - 2 + kd;
            if (id < 0 || id >= 8) continue;
            for (int kh = 0; kh < 5; ++kh) {
                const int ih = oh - 2 + kh;
                if (ih < 0 || ih >= 8) continue;
                #pragma unroll
                for (int kw = 0; kw < 5; ++kw) {
                    const int iw = ow - 2 + kw;
                    if (iw < 0 || iw >= 8) continue;
                    acc += ib[id * 64 + ih * 8 + iw] * wc[(kd * 5 + kh) * 5 + kw];
                }
            }
        }
    }
    out[o] = acc;
}

// BatchNorm (training-mode batch statistics over B and spatial dims) + ELU.
// One block per channel; 2048 values per channel.
__global__ void __launch_bounds__(256)
bnelu_kernel(const float* __restrict__ raw, const float* __restrict__ g,
             const float* __restrict__ beta, float* __restrict__ out)
{
    const int c = blockIdx.x;
    const int t = threadIdx.x;
    __shared__ float ssum[256];
    __shared__ float ssq[256];

    float v[8];
    float s = 0.0f, q = 0.0f;
    #pragma unroll
    for (int j = 0; j < 8; ++j) {
        const int lin = j * 256 + t;        // 0..2047
        const int b   = lin >> 9;
        const int sp  = lin & 511;
        const float x = raw[(b * 8 + c) * 512 + sp];
        v[j] = x; s += x; q += x * x;
    }
    ssum[t] = s; ssq[t] = q;
    __syncthreads();
    for (int off = 128; off > 0; off >>= 1) {
        if (t < off) { ssum[t] += ssum[t + off]; ssq[t] += ssq[t + off]; }
        __syncthreads();
    }
    const float mean  = ssum[0] * (1.0f / 2048.0f);
    const float var   = ssq[0] * (1.0f / 2048.0f) - mean * mean;
    const float scale = rsqrtf(var + 1e-5f) * g[c];
    const float shift = beta[c];

    #pragma unroll
    for (int j = 0; j < 8; ++j) {
        const int lin = j * 256 + t;
        const int b   = lin >> 9;
        const int sp  = lin & 511;
        out[(b * 8 + c) * 512 + sp] = eluf((v[j] - mean) * scale + shift);
    }
}

// ------------------------------- launcher ----------------------------------
extern "C" void kernel_launch(void* const* d_in, const int* in_sizes, int n_in,
                              void* d_out, int out_size, void* d_ws, size_t ws_size,
                              hipStream_t stream)
{
    const float* goals       = (const float*)d_in[0];
    const float* inputs      = (const float*)d_in[1];
    const float* backgrounds = (const float*)d_in[2];
    const float* W1 = (const float*)d_in[3];
    const float* b1 = (const float*)d_in[4];
    const float* W2 = (const float*)d_in[5];
    const float* b2 = (const float*)d_in[6];
    const float* W3 = (const float*)d_in[7];
    const float* b3 = (const float*)d_in[8];
    const float* conv1_w = (const float*)d_in[9];
    const float* conv1_b = (const float*)d_in[10];
    const float* bn1_g   = (const float*)d_in[11];
    const float* bn1_b   = (const float*)d_in[12];
    const float* convs_w = (const float*)d_in[13];
    const float* convs_b = (const float*)d_in[14];
    const float* bns_g   = (const float*)d_in[15];
    const float* bns_b   = (const float*)d_in[16];

    float* out  = (float*)d_out;
    float* ws   = (float*)d_ws;
    float* grid = ws;                        // SCENE_SZ floats
    float* raw  = ws + SCENE_SZ;             // ACT_SZ floats
    float* act  = ws + SCENE_SZ + ACT_SZ;    // ACT_SZ floats

    // 1) zero the scene grid (scatter-max base; reference starts from zeros)
    zero_kernel<<<(SCENE_SZ + 255) / 256, 256, 0, stream>>>(grid, SCENE_SZ);

    // 2) point MLP + scatter-max, 3 clouds via gridDim.y
    dim3 egrid(192, 3, 1);
    encode_kernel<<<egrid, 256, 0, stream>>>(inputs, goals, backgrounds,
                                             W1, b1, W2, b2, W3, b3, grid);

    // 3) conv stack: conv -> BN(batch stats) -> ELU, 7 layers
    conv1_kernel<<<ACT_SZ / 256, 256, 0, stream>>>(grid, conv1_w, conv1_b, raw);
    bnelu_kernel<<<8, 256, 0, stream>>>(raw, bn1_g, bn1_b, act);
    for (int i = 0; i < 6; ++i) {
        float* dst = (i == 5) ? out : act;
        convN_kernel<<<ACT_SZ / 256, 256, 0, stream>>>(
            act, convs_w + i * 8 * 8 * 125, convs_b + i * 8, raw);
        bnelu_kernel<<<8, 256, 0, stream>>>(raw, bns_g + i * 8, bns_b + i * 8, dst);
    }
}
